// G8Linear_24721831756570
// MI455X (gfx1250) — matrix operand order repaired and checked by reading, not compiled, not run on hardware
//
#include <hip/hip_runtime.h>
#include <hip/hip_bf16.h>
#include <stdint.h>

typedef __attribute__((ext_vector_type(16))) _Float16 v16h;
typedef __attribute__((ext_vector_type(2)))  _Float16 h2;
typedef __attribute__((ext_vector_type(8)))  float    v8f;
typedef __attribute__((ext_vector_type(4)))  unsigned int u32x4;
typedef __attribute__((ext_vector_type(4)))  int      i32x4;
typedef __attribute__((ext_vector_type(8)))  int      i32x8;

#define TM 64        // M tile (full M=64)
#define TN 32        // O tile per block
#define TK 128       // K tile per outer iteration (4 WMMA k-steps)
#define LDW 65       // LDS row stride in dwords (64 data + 1 pad -> odd, conflict-free)
#define ABUF_DW (TM * LDW)   // one sA buffer: 64*65 = 4160 dw = 16640 B
#define BBUF_DW (TN * LDW)   // one sB buffer: 32*65 = 2080 dw =  8320 B

// ---------------------------------------------------------------------------
// TDM: 2D tensor load of a TM x TK f16 tile (row stride = K elements) from
// global into LDS, HW-padding 1 dword per 64 dwords => LDS row stride 65 dw.
// ---------------------------------------------------------------------------
__device__ __forceinline__ void tdm_issue_xtile(uint32_t lds_byte,
                                                const _Float16* gsrc, int K) {
    uint64_t ga = (uint64_t)(uintptr_t)gsrc;
    u32x4 g0;
    g0[0] = 1u;                                   // count=1, user descriptor
    g0[1] = lds_byte;                             // lds_addr (bytes)
    g0[2] = (uint32_t)ga;                         // global_addr[31:0]
    g0[3] = (uint32_t)((ga >> 32) & 0x1FFFFFFu)   // global_addr[56:32]
          | (2u << 30);                           // type = 2 ("image")
    i32x8 g1;
    g1[0] = (1 << 16)                             // data_size = 2 bytes
          | (1 << 20)                             // pad_enable
          | (5 << 22);                            // pad_interval: 64 dwords
                                                  // pad_amount = 0 -> 1 dword
    g1[1] = (int)((unsigned)(K & 0xFFFF) << 16);  // tensor_dim0[15:0] @ [31:16]
    g1[2] = (int)(((unsigned)K >> 16)             // tensor_dim0[31:16]
          | ((unsigned)TM << 16));                // tensor_dim1[15:0] = 64 rows
    g1[3] = (int)((unsigned)TK << 16);            // tile_dim0 = 128 elements
    g1[4] = TM;                                   // tile_dim1 = 64 rows, tile_dim2=0
    g1[5] = K;                                    // tensor_dim0_stride[31:0]
    g1[6] = 0;
    g1[7] = 0;
    i32x4 gz = {};
#if __clang_major__ >= 23
    i32x8 gz8 = {};
    __builtin_amdgcn_tensor_load_to_lds(g0, g1, gz, gz, gz8, 0);
#else
    __builtin_amdgcn_tensor_load_to_lds(g0, g1, gz, gz, 0);
#endif
}

__global__ __launch_bounds__(256) void g8linear_wmma(
    const float*    __restrict__ x,        // [M,K] f32
    const int*      __restrict__ qw,       // [O,K/2] one packed byte per int32
    const float*    __restrict__ scales,   // [O,K/8]
    const float*    __restrict__ offsets,  // [O,K/8]
    const float*    __restrict__ bias,     // [O]
    const _Float16* __restrict__ xh,       // [M,K] f16 (pre-converted, in ws)
    float*          __restrict__ y,        // [M,O]
    int K, int O, int use_tdm)
{
    __shared__ uint32_t sA[3 * ABUF_DW];          // triple-buffered x tile
    __shared__ uint32_t sB[2 * BBUF_DW];          // double-buffered W tile

    const int t    = threadIdx.x;
    const int lane = t & 31;
    const int wave = t >> 5;                      // 0..7
    const int o0   = blockIdx.x * TN;

    const int mBase  = (wave >> 1) * 16;          // 0,16,32,48
    const int nBase  = (wave & 1) * 16;           // 0,16
    const int half16 = lane >> 4;                 // 0 or 1
    const int l16    = lane & 15;

    const int Khalf = K >> 1;                     // packed bytes (int32s) per O row
    const int G     = K >> 3;                     // groups per O row
    const int nIter = K / TK;

    const uint32_t ldsA0 = (uint32_t)(uintptr_t)&sA[0];

    // dequant mapping: thread handles 2 quads; quad qi -> row r, quad-col c4.
    // 4 packed bytes = 8 k values = exactly one group of 8.
    const int qRow0 = t >> 4;                     // quads 0..255: r = t/16
    const int qCol0 = t & 15;                     // c4 = t%16
    // second quad: qi = t + 256 -> r = qRow0 + 16, same c4

    v8f acc = {};

    if (use_tdm && wave == 0) {                   // prefetch tile 0 into buf 0
        tdm_issue_xtile(ldsA0, xh, K);
    }

    int bufA = 0;                                 // it % 3
    for (int it = 0; it < nIter; ++it) {
        const int k0   = it * TK;
        const int bufB = it & 1;
        const int nxtA = (bufA == 2) ? 0 : bufA + 1;

        if (use_tdm) {
            if (wave == 0 && it + 1 < nIter) {    // prefetch next tile
                tdm_issue_xtile(ldsA0 + (uint32_t)nxtA * (ABUF_DW * 4),
                                xh + (it + 1) * TK, K);
            }
        } else {
            // manual staging: TM x TK f32 -> packed f16 pairs (64 pairs/row)
            #pragma unroll
            for (int j = 0; j < (TM * (TK / 2)) / 256; ++j) {
                int p = j * 256 + t;
                int r = p >> 6;                   // 64 pairs per row
                int c = p & 63;
                h2 v;
                v.x = (_Float16)x[r * K + k0 + 2 * c];
                v.y = (_Float16)x[r * K + k0 + 2 * c + 1];
                sA[bufA * ABUF_DW + r * LDW + c] = __builtin_bit_cast(uint32_t, v);
            }
        }

        // ---- dequantize W tile: 2 quads (2x b128) per thread ----
        #pragma unroll
        for (int j = 0; j < 2; ++j) {
            int r  = qRow0 + j * 16;              // 0..31
            int o  = o0 + r;
            int c4 = qCol0;                       // 0..15
            const i32x4* src =
                (const i32x4*)(qw + o * Khalf + (k0 >> 1) + 4 * c4);
            i32x4 pk4 = *src;                     // 4 packed bytes (b128)
            int g = (k0 >> 3) + c4;               // one group per quad
            _Float16 s   = (_Float16)scales[o * G + g];
            _Float16 off = (_Float16)offsets[o * G + g];
            h2 sv; sv.x = s;   sv.y = s;
            h2 ov; ov.x = off; ov.y = off;
            uint32_t base = (uint32_t)(bufB * BBUF_DW + r * LDW + 4 * c4);
            #pragma unroll
            for (int e = 0; e < 4; ++e) {
                int pk = pk4[e] & 0xFF;
                h2 q; q.x = (_Float16)(pk & 15); q.y = (_Float16)((pk >> 4) & 15);
                h2 w = q * sv + ov;               // v_pk_fma_f16
                sB[base + e] = __builtin_bit_cast(uint32_t, w);
            }
            if (it + 1 < nIter) {                 // stream hint for next tile
                __builtin_prefetch(qw + o * Khalf + ((k0 + TK) >> 1) + 4 * c4, 0, 1);
            }
        }

        if (use_tdm && wave == 0) {
            if (it + 1 < nIter) __builtin_amdgcn_s_wait_tensorcnt(1);
            else                __builtin_amdgcn_s_wait_tensorcnt(0);
        }
        __syncthreads();                          // single barrier per iteration

        // ---- four 16x16x32 f16 WMMA steps across this K tile ----
        #pragma unroll
        for (int kk2 = 0; kk2 < TK / 2; kk2 += 16) {   // dword offset per k-slice
            union { v16h v; uint32_t u[8]; } af, bf;
            int am = mBase + l16;
            #pragma unroll
            for (int v = 0; v < 8; ++v) {
                // A 16-bit 16x32: VGPR v<4 -> K octet 0, v>=4 -> +16 halves;
                // lane half adds +8 halves; dword offset = half offset / 2
                int kd = ((v < 4) ? 0 : 8) + 4 * half16 + (v & 3);
                af.u[v] = sA[bufA * ABUF_DW + am * LDW + kk2 + kd];
            }
            int bn = nBase + l16;
            #pragma unroll
            for (int v = 0; v < 8; ++v) {
                // B 16-bit 32x16: same K order as A (VGPR v<4 -> K octet 0, v>=4 -> +16; lane half +8)
                bf.u[v] = sB[bufB * BBUF_DW + bn * LDW + kk2 + ((v < 4) ? 0 : 8) + 4 * half16 + (v & 3)];
            }
            acc = __builtin_amdgcn_wmma_f32_16x16x32_f16(
                false, af.v, false, bf.v, (short)0, acc, false, false);
        }
        bufA = nxtA;
    }

    // ---- epilogue: C/D layout (VGPR r: M = mBase + r + 8*half16, N = nBase + l16)
    int n = o0 + nBase + l16;
    float b = bias[n];
    #pragma unroll
    for (int r = 0; r < 8; ++r) {
        int m = mBase + r + 8 * half16;
        y[m * O + n] = acc[r] + b;
    }
}

// ---- pre-pass: convert x (f32) to f16 pairs in workspace ----
__global__ __launch_bounds__(256) void cvt_x_f16(const float* __restrict__ x,
                                                 uint32_t* __restrict__ xh,
                                                 int nPairs) {
    int stride = gridDim.x * blockDim.x;
    for (int p = blockIdx.x * blockDim.x + threadIdx.x; p < nPairs; p += stride) {
        h2 v;
        v.x = (_Float16)x[2 * p];
        v.y = (_Float16)x[2 * p + 1];
        xh[p] = __builtin_bit_cast(uint32_t, v);
    }
}

extern "C" void kernel_launch(void* const* d_in, const int* in_sizes, int n_in,
                              void* d_out, int out_size, void* d_ws, size_t ws_size,
                              hipStream_t stream) {
    const float* x       = (const float*)d_in[0];
    const int*   qw      = (const int*)d_in[1];
    const float* scales  = (const float*)d_in[2];
    const float* offsets = (const float*)d_in[3];
    const float* bias    = (const float*)d_in[4];
    float*       y       = (float*)d_out;

    const int M = 64;
    const int K = in_sizes[0] / M;   // 4096
    const int O = in_sizes[4];       // 11008

    const size_t need = (size_t)M * (size_t)K * sizeof(_Float16);  // 512 KB
    const int use_tdm = (ws_size >= need) ? 1 : 0;
    _Float16* xh = (_Float16*)d_ws;

    if (use_tdm) {
        int nPairs = (M * K) / 2;
        hipLaunchKernelGGL(cvt_x_f16, dim3(256), dim3(256), 0, stream,
                           x, (uint32_t*)d_ws, nPairs);
    }

    dim3 grid(O / TN);               // 344 blocks
    dim3 block(256);                 // 8 waves (wave32)
    hipLaunchKernelGGL(g8linear_wmma, grid, block, 0, stream,
                       x, qw, scales, offsets, bias, xh, y, K, O, use_tdm);
}
